// SelfAttention_83691732730308
// MI455X (gfx1250) — matrix operand order repaired and checked by reading, not compiled, not run on hardware
//
#include <hip/hip_runtime.h>

typedef __attribute__((ext_vector_type(16))) __bf16 v16bf;
typedef __attribute__((ext_vector_type(8)))  __bf16 v8bf;
typedef __attribute__((ext_vector_type(8)))  float  v8f;
typedef int v4i32 __attribute__((vector_size(16)));

#define AS1 __attribute__((address_space(1)))
#define AS3 __attribute__((address_space(3)))

#if defined(__gfx1250__) && __has_builtin(__builtin_amdgcn_global_load_async_to_lds_b128)
#define HAVE_ASYNC_LDS 1
#else
#define HAVE_ASYNC_LDS 0
#endif

static __device__ inline void wait_async_all() {
#if HAVE_ASYNC_LDS
#if __has_builtin(__builtin_amdgcn_s_wait_asynccnt)
  __builtin_amdgcn_s_wait_asynccnt(0);
#else
  asm volatile("s_wait_asynccnt 0" ::: "memory");
#endif
#endif
}

constexpr int Bb = 2, Ss = 2048, Dd = 1024, Hh = 16, HDm = 64;

static __device__ inline v8f zerov8() {
  v8f z;
#pragma unroll
  for (int i = 0; i < 8; ++i) z[i] = 0.0f;
  return z;
}

// A-fragment (16x32 bf16, M x K): lane L -> M = L%16, half = L/16.
// K(e) = 16*half + e: one contiguous 32B run.
static __device__ inline v16bf load_a_frag(const __bf16* __restrict__ base, int ld,
                                           int m, int half) {
  const __bf16* p = base + m * ld + 16 * half;
  v8bf lo = *(const v8bf*)p;
  v8bf hi = *(const v8bf*)(p + 8);
  return __builtin_shufflevector(lo, hi, 0, 1, 2, 3, 4, 5, 6, 7,
                                 8, 9, 10, 11, 12, 13, 14, 15);
}

// B-fragment (32x16 bf16, K x N) from a row-major [N x K] matrix (K contiguous):
// lane L -> N = L%16, half = L/16, K(e) = 16*half + e: one contiguous 32B run.
static __device__ inline v16bf load_b_frag(const __bf16* __restrict__ base, int ld,
                                           int n, int half) {
  const __bf16* p = base + n * ld + 16 * half;
  v8bf lo = *(const v8bf*)p;
  v8bf hi = *(const v8bf*)(p + 8);
  return __builtin_shufflevector(lo, hi, 0, 1, 2, 3, 4, 5, 6, 7,
                                 8, 9, 10, 11, 12, 13, 14, 15);
}

static __device__ inline v8f wmma_bf16(v16bf a, v16bf b, v8f c) {
  return __builtin_amdgcn_wmma_f32_16x16x32_bf16(false, a, false, b,
                                                 (short)0, c, false, false);
}

__global__ void cvt_bf16(const float* __restrict__ in, __bf16* __restrict__ out, int n) {
  int i = blockIdx.x * blockDim.x + threadIdx.x;
  int stride = gridDim.x * blockDim.x;
  for (; i < n; i += stride) out[i] = (__bf16)in[i];
}

// Y = X @ W^T  (X: [M,K] bf16 row-major, W: [N,K] bf16 row-major)
// MODE 0: Y row-major [M,N] of OutT (+ optional fp32 bias over N)
// MODE 1: per-head transposed store for V: Y[((b*H + h)*HD + d)*S + s]
// Double-buffered LDS pipeline; async global->LDS loads when available.
template <typename OutT, int MODE, bool BIAS>
__global__ __launch_bounds__(256) void gemm_xwt(const __bf16* __restrict__ X,
                                                const __bf16* __restrict__ W,
                                                OutT* __restrict__ Y,
                                                const float* __restrict__ bias,
                                                int M, int N, int K) {
  __shared__ __align__(16) __bf16 As[2][128 * 32];
  __shared__ __align__(16) __bf16 Bs[2][128 * 32];
  const int tid = threadIdx.x;
  const int lane = tid & 31, wid = tid >> 5;
  const int half = lane >> 4, l16 = lane & 15;
  const int waveM = wid & 3, waveN = wid >> 2;  // 4 (M) x 2 (N) waves
  const int m_blk = blockIdx.y * 128;
  const int n_blk = blockIdx.x * 128;

  auto load_tile = [&](int k0, int buf) {
#pragma unroll
    for (int i = 0; i < 2; ++i) {  // 512 chunks of 8 bf16 (16B) per tile
      int c = tid + i * 256;
      int row = c >> 2, q = c & 3;
      const __bf16* gA = &X[(long)(m_blk + row) * K + k0 + q * 8];
      const __bf16* gB = &W[(long)(n_blk + row) * K + k0 + q * 8];
      __bf16* lA = &As[buf][row * 32 + q * 8];
      __bf16* lB = &Bs[buf][row * 32 + q * 8];
#if HAVE_ASYNC_LDS
      __builtin_amdgcn_global_load_async_to_lds_b128((v4i32*)gA,
                                                     (AS3 v4i32*)lA, 0, 0);
      __builtin_amdgcn_global_load_async_to_lds_b128((v4i32*)gB,
                                                     (AS3 v4i32*)lB, 0, 0);
#else
      *(uint4*)lA = *(const uint4*)gA;
      *(uint4*)lB = *(const uint4*)gB;
#endif
    }
  };

  v8f acc[2][4];
#pragma unroll
  for (int s = 0; s < 2; ++s)
#pragma unroll
    for (int f = 0; f < 4; ++f) acc[s][f] = zerov8();

  const int NT = K >> 5;  // K/32 tiles
  load_tile(0, 0);
  int buf = 0;
  for (int t = 0; t < NT; ++t) {
    wait_async_all();
    __syncthreads();  // tile t resident; previous reads of buf^1 done
    if (t + 1 < NT) load_tile((t + 1) << 5, buf ^ 1);

    v16bf a[2], b[4];
#pragma unroll
    for (int s = 0; s < 2; ++s)
      a[s] = load_a_frag(As[buf], 32, waveM * 32 + s * 16 + l16, half);
#pragma unroll
    for (int f = 0; f < 4; ++f)
      b[f] = load_b_frag(Bs[buf], 32, waveN * 64 + f * 16 + l16, half);
#pragma unroll
    for (int s = 0; s < 2; ++s)
#pragma unroll
      for (int f = 0; f < 4; ++f) acc[s][f] = wmma_bf16(a[s], b[f], acc[s][f]);
    buf ^= 1;
  }

#pragma unroll
  for (int s = 0; s < 2; ++s)
#pragma unroll
    for (int f = 0; f < 4; ++f) {
      int n = n_blk + waveN * 64 + f * 16 + l16;
#pragma unroll
      for (int r = 0; r < 8; ++r) {
        int m = m_blk + waveM * 32 + s * 16 + r + 8 * half;
        float v = acc[s][f][r];
        if (BIAS) v += bias[n];
        if (MODE == 0) {
          Y[(long)m * N + n] = (OutT)v;
        } else {
          int bb = m >> 11, srow = m & (Ss - 1);  // S = 2048
          int hh = n >> 6, dd = n & (HDm - 1);    // HD = 64
          Y[(long)((bb * Hh + hh) * HDm + dd) * Ss + srow] = (OutT)v;
        }
      }
    }
}

// Flash attention: one wave handles 16 query rows of one (b, h).
__global__ __launch_bounds__(128) void attn_kernel(const __bf16* __restrict__ Q,
                                                   const __bf16* __restrict__ Kc,
                                                   const __bf16* __restrict__ Vt,
                                                   __bf16* __restrict__ ctx) {
  __shared__ __align__(16) __bf16 Pt[4][16 * 32];
  const int tid = threadIdx.x;
  const int lane = tid & 31, wid = tid >> 5;
  const int half = lane >> 4, l16 = lane & 15;
  const int gw = blockIdx.x * 4 + wid;
  const int qb = gw & 127;         // S/16 = 128 q-blocks
  const int h = (gw >> 7) & 15;    // H = 16
  const int b = gw >> 11;

  const __bf16* Qbase = Q + (long)(b * Ss + qb * 16) * Dd + h * HDm;
  v16bf q0 = load_a_frag(Qbase, Dd, l16, half);        // d = 0..31
  v16bf q1 = load_a_frag(Qbase + 32, Dd, l16, half);   // d = 32..63

  const __bf16* Krow0 = Kc + (long)(b * Ss) * Dd + h * HDm;        // + j*Dd
  const __bf16* Vrow0 = Vt + (long)((b * Hh + h) * HDm) * Ss;      // + d*Ss + j

  float mi[8], li[8];
  v8f o[4];
#pragma unroll
  for (int r = 0; r < 8; ++r) { mi[r] = -1e30f; li[r] = 0.0f; }
#pragma unroll
  for (int f = 0; f < 4; ++f) o[f] = zerov8();

  const float scale = 0.125f;  // 1/sqrt(64)

  for (int j0 = 0; j0 < Ss; j0 += 32) {
    // Prefetch next kv tile (K rows j0+32..j0+63; V columns at j0+32).
    int jn = j0 + 32;
    if (jn < Ss) {
      __builtin_prefetch(Krow0 + (long)(jn + lane) * Dd, 0, 3);
      __builtin_prefetch(Vrow0 + (long)(lane * 2) * Ss + jn, 0, 3);
    }
    // S = Q K^T for 16 x 32 score tile (two 16x16 frags, chained over d)
    v8f sc[2];
#pragma unroll
    for (int f = 0; f < 2; ++f) {
      const __bf16* Kbase = Krow0 + (long)(j0 + f * 16) * Dd;
      v16bf b0 = load_b_frag(Kbase, Dd, l16, half);
      v16bf b1 = load_b_frag(Kbase + 32, Dd, l16, half);
      v8f c = zerov8();
      c = wmma_bf16(q0, b0, c);
      c = wmma_bf16(q1, b1, c);
      sc[f] = c;
    }
    // Online softmax. C/D layout: row M = r + 8*half lives in this 16-lane
    // group -> row reductions are shfl_xor over masks 1,2,4,8.
    float alpha[8];
#pragma unroll
    for (int r = 0; r < 8; ++r) {
      float rm = fmaxf(sc[0][r], sc[1][r]);
#pragma unroll
      for (int msk = 1; msk < 16; msk <<= 1)
        rm = fmaxf(rm, __shfl_xor(rm, msk, 32));
      rm *= scale;
      float mnew = fmaxf(mi[r], rm);
      alpha[r] = __expf(mi[r] - mnew);
      float p0 = __expf(sc[0][r] * scale - mnew);
      float p1 = __expf(sc[1][r] * scale - mnew);
      Pt[wid][(r + 8 * half) * 32 + l16] = (__bf16)p0;
      Pt[wid][(r + 8 * half) * 32 + 16 + l16] = (__bf16)p1;
      float s2 = p0 + p1;
#pragma unroll
      for (int msk = 1; msk < 16; msk <<= 1) s2 += __shfl_xor(s2, msk, 32);
      li[r] = li[r] * alpha[r] + s2;
      mi[r] = mnew;
    }
#pragma unroll
    for (int f = 0; f < 4; ++f)
#pragma unroll
      for (int r = 0; r < 8; ++r) o[f][r] *= alpha[r];

    __builtin_amdgcn_wave_barrier();  // LDS bounce: C/D layout -> A layout
    v16bf pa = load_a_frag(Pt[wid], 32, l16, half);
#pragma unroll
    for (int f = 0; f < 4; ++f) {
      const __bf16* Vb = Vrow0 + (long)(f * 16 + l16) * Ss + j0 + 16 * half;
      v8bf lo = *(const v8bf*)Vb;
      v8bf hi = *(const v8bf*)(Vb + 8);
      v16bf bv = __builtin_shufflevector(lo, hi, 0, 1, 2, 3, 4, 5, 6, 7,
                                         8, 9, 10, 11, 12, 13, 14, 15);
      o[f] = wmma_bf16(pa, bv, o[f]);
    }
    __builtin_amdgcn_wave_barrier();
  }

#pragma unroll
  for (int f = 0; f < 4; ++f)
#pragma unroll
    for (int r = 0; r < 8; ++r) {
      int row = qb * 16 + r + 8 * half;
      int col = h * HDm + f * 16 + l16;
      ctx[(long)(b * Ss + row) * Dd + col] = (__bf16)(o[f][r] / li[r]);
    }
}

extern "C" void kernel_launch(void* const* d_in, const int* in_sizes, int n_in,
                              void* d_out, int out_size, void* d_ws, size_t ws_size,
                              hipStream_t stream) {
  const float* x  = (const float*)d_in[0];
  const float* Wq = (const float*)d_in[1];
  const float* Wk = (const float*)d_in[2];
  const float* Wv = (const float*)d_in[3];
  const float* Wo = (const float*)d_in[4];
  const float* bo = (const float*)d_in[5];
  float* out = (float*)d_out;

  const size_t nX = (size_t)Bb * Ss * Dd;  // 4M elements
  const size_t nW = (size_t)Dd * Dd;       // 1M elements

  char* ws = (char*)d_ws;
  __bf16* xb  = (__bf16*)ws; ws += nX * 2;
  __bf16* wqb = (__bf16*)ws; ws += nW * 2;
  __bf16* wkb = (__bf16*)ws; ws += nW * 2;
  __bf16* wvb = (__bf16*)ws; ws += nW * 2;
  __bf16* wob = (__bf16*)ws; ws += nW * 2;
  __bf16* Qb  = (__bf16*)ws; ws += nX * 2;
  __bf16* Kb  = (__bf16*)ws; ws += nX * 2;
  __bf16* Vtb = (__bf16*)ws; ws += nX * 2;
  __bf16* ctx = (__bf16*)ws; ws += nX * 2;

  cvt_bf16<<<512, 256, 0, stream>>>(x, xb, (int)nX);
  cvt_bf16<<<256, 256, 0, stream>>>(Wq, wqb, (int)nW);
  cvt_bf16<<<256, 256, 0, stream>>>(Wk, wkb, (int)nW);
  cvt_bf16<<<256, 256, 0, stream>>>(Wv, wvb, (int)nW);
  cvt_bf16<<<256, 256, 0, stream>>>(Wo, wob, (int)nW);

  dim3 gg(Dd / 128, (Bb * Ss) / 128);  // (8, 32)
  gemm_xwt<__bf16, 0, false><<<gg, 256, 0, stream>>>(xb, wqb, Qb, nullptr,
                                                     Bb * Ss, Dd, Dd);
  gemm_xwt<__bf16, 0, false><<<gg, 256, 0, stream>>>(xb, wkb, Kb, nullptr,
                                                     Bb * Ss, Dd, Dd);
  gemm_xwt<__bf16, 1, false><<<gg, 256, 0, stream>>>(xb, wvb, Vtb, nullptr,
                                                     Bb * Ss, Dd, Dd);

  attn_kernel<<<(Bb * Hh * (Ss / 16)) / 4, 128, 0, stream>>>(Qb, Kb, Vtb, ctx);

  gemm_xwt<float, 0, true><<<gg, 256, 0, stream>>>(ctx, wob, out, bo,
                                                   Bb * Ss, Dd, Dd);
}
